// MGAE_6631429505271
// MI455X (gfx1250) — compile-verified, matrix operand order checked by reading, not run on hardware
//
#include <hip/hip_runtime.h>
#include <stdint.h>

// ---------------- problem constants (reference shapes are fixed) -----------
#define GNODES 4096
#define GEDGES 65536
#define GIN    512
#define GH0    128
#define GH1    64
#define GNH    2048   // N/2 for GFN hidden

// ---------------- WMMA GEMM tile config ------------------------------------
// Invariants: every GEMM here has M % BM == 0 and K % BK == 0; only N guarded.
#define BM 128
#define BN 128
#define BK 32

typedef __attribute__((ext_vector_type(16))) __bf16 v16bf;
typedef __attribute__((ext_vector_type(8)))  __bf16 v8bf;
typedef __attribute__((ext_vector_type(8)))  float  v8f;

__device__ __forceinline__ unsigned short f2bf(float f) {
  union { float f; unsigned u; } c; c.f = f;
  unsigned u = c.u;
  u += 0x7FFFu + ((u >> 16) & 1u);         // round-to-nearest-even
  return (unsigned short)(u >> 16);
}
__device__ __forceinline__ unsigned pack2(unsigned short a, unsigned short b) {
  return (unsigned)a | ((unsigned)b << 16);
}

// ---------------------------------------------------------------------------
// C = act( beta*C + (A*arow) @ (B*brow) + bias ), bf16 WMMA, f32 accumulate.
// AMODE: 0 = A f32 row-scaled by arow   1 = A u8 bitmask -> popcount
//        2 = A raw bf16
// BMODE: 0 = B f32 [k][n] (col-scaled by brow[k])
//        1 = B f32 [n][k] (transposed source, for h @ h^T)
//        2 = B bf16 [n][k] pre-transposed (zero-conversion staging)
// arow/brow/bias are ALWAYS valid (ones[]/zeros[] defaults) -> branchless.
// Double-buffered LDS, one barrier per K step, b128/b64 staging.
// ---------------------------------------------------------------------------
template<int AMODE, int BMODE>
__global__ __launch_bounds__(256) void gemm_wmma(
    const void* __restrict__ Ap, const void* __restrict__ Bp, float* __restrict__ C,
    int N, int K, int lda, int ldb, int ldc,
    const float* __restrict__ arow, const float* __restrict__ brow,
    const float* __restrict__ bias,
    int beta, int act)
{
  __shared__ __align__(16) unsigned short lA[2][BM * BK];   // [m][k]
  __shared__ __align__(16) unsigned short lB[2][BN * BK];   // [n][k]

  const int tid  = threadIdx.x;
  const int lane = tid & 31;
  const int wave = tid >> 5;
  const int wm   = (wave & 3) * 32;   // 4 waves down M
  const int wn   = (wave >> 2) * 64;  // 2 waves across N
  const int bm   = blockIdx.y * BM;
  const int bn   = blockIdx.x * BN;

  const float* Bf = (const float*)Bp;
  unsigned       regA[4][2];
  unsigned       regB[4][2];    // BMODE 1/2
  unsigned short regBs[16];     // BMODE 0

  auto loadA = [&](int k0) {
    const bool pf = (k0 + BK) < K;
#pragma unroll
    for (int i = 0; i < 4; ++i) {
      int q = tid + i * 256;
      int r = q >> 3, c = (q & 7) * 4;
      if (AMODE == 0) {
        const float4 v = *(const float4*)((const float*)Ap + (size_t)(bm + r) * lda + (k0 + c));
        float s = arow[bm + r];
        regA[i][0] = pack2(f2bf(v.x * s), f2bf(v.y * s));
        regA[i][1] = pack2(f2bf(v.z * s), f2bf(v.w * s));
      } else if (AMODE == 1) {
        unsigned b = *(const unsigned*)((const unsigned char*)Ap + (size_t)(bm + r) * lda + (k0 + c));
        regA[i][0] = pack2(f2bf((float)__popc((int)(b & 7u))),
                           f2bf((float)__popc((int)((b >> 8) & 7u))));
        regA[i][1] = pack2(f2bf((float)__popc((int)((b >> 16) & 7u))),
                           f2bf((float)__popc((int)((b >> 24) & 7u))));
      } else {
        const uint2 v = *(const uint2*)((const unsigned short*)Ap + (size_t)(bm + r) * lda + (k0 + c));
        regA[i][0] = v.x; regA[i][1] = v.y;
      }
      if (pf)   // L2 prefetch of tile after next (speculative-safe)
        __builtin_prefetch((const char*)Ap + ((size_t)(bm + r) * lda + (k0 + BK + c)) *
                               (AMODE == 0 ? 4 : (AMODE == 1 ? 1 : 2)), 0, 1);
    }
  };
  auto storeA = [&](int buf) {
#pragma unroll
    for (int i = 0; i < 4; ++i) {
      int q = tid + i * 256;
      int r = q >> 3, c = (q & 7) * 4;
      *(uint2*)&lA[buf][r * BK + c] = make_uint2(regA[i][0], regA[i][1]);
    }
  };

  auto loadB = [&](int k0) {
#pragma unroll
    for (int i = 0; i < 4; ++i) {
      int q = tid + i * 256;
      if (BMODE == 2) {                      // bf16 [n][k]: pure copy (full tiles)
        int n = q >> 3, c = (q & 7) * 4;
        const uint2 v = *(const uint2*)((const unsigned short*)Bp + (size_t)(bn + n) * ldb + (k0 + c));
        regB[i][0] = v.x; regB[i][1] = v.y;
        if ((k0 + BK) < K)
          __builtin_prefetch((const char*)Bp + ((size_t)(bn + n) * ldb + (k0 + BK + c)) * 2, 0, 1);
      } else if (BMODE == 1) {               // f32 [n][k] (h @ h^T)
        int n = q >> 3, c = (q & 7) * 4;
        int gn = bn + n;
        int gns = min(gn, N - 1);            // clamped address, always in-bounds
        const float4 v = *(const float4*)(Bf + (size_t)gns * ldb + (k0 + c));
        float g = (gn < N) ? 1.f : 0.f;      // guard folded into scale (branchless)
        regB[i][0] = pack2(f2bf(v.x * brow[k0+c]   * g), f2bf(v.y * brow[k0+c+1] * g));
        regB[i][1] = pack2(f2bf(v.z * brow[k0+c+2] * g), f2bf(v.w * brow[k0+c+3] * g));
      } else {                               // f32 [k][n]
        int kk = q >> 5, n4 = (q & 31) * 4;
        int gn = bn + n4;
        int gns = min(gn, N - 4);            // clamped address, always in-bounds
        const float4 v = *(const float4*)(Bf + (size_t)(k0 + kk) * ldb + gns);
        float s = (gn < N) ? brow[k0 + kk] : 0.f;   // guard folded into scale
        regBs[i * 4 + 0] = f2bf(v.x * s);
        regBs[i * 4 + 1] = f2bf(v.y * s);
        regBs[i * 4 + 2] = f2bf(v.z * s);
        regBs[i * 4 + 3] = f2bf(v.w * s);
      }
    }
  };
  auto storeB = [&](int buf) {
#pragma unroll
    for (int i = 0; i < 4; ++i) {
      int q = tid + i * 256;
      if (BMODE != 0) {
        int n = q >> 3, c = (q & 7) * 4;
        *(uint2*)&lB[buf][n * BK + c] = make_uint2(regB[i][0], regB[i][1]);
      } else {
        int kk = q >> 5, n4 = (q & 31) * 4;
#pragma unroll
        for (int j = 0; j < 4; ++j) lB[buf][(n4 + j) * BK + kk] = regBs[i * 4 + j];
      }
    }
  };

  v8f acc[2][4];
#pragma unroll
  for (int mi = 0; mi < 2; ++mi)
#pragma unroll
    for (int ni = 0; ni < 4; ++ni)
#pragma unroll
      for (int r = 0; r < 8; ++r) acc[mi][ni][r] = 0.f;

  loadA(0); loadB(0);
  storeA(0); storeB(0);

  int buf = 0;
  for (int k0 = 0; k0 < K; k0 += BK, buf ^= 1) {
    __syncthreads();                       // lA/lB[buf] ready; lA/lB[buf^1] free
    const bool more = (k0 + BK) < K;
    if (more) { loadA(k0 + BK); loadB(k0 + BK); }  // global loads fly over compute

    // fragments per ISA 16-bit 16x32 layout:
    // lane l: row/col = l&15; K chunks [h8, h8+8) U [16+h8, 16+h8+8), h8=(l>>4)*8
    const int rc = lane & 15;
    const int h8 = (lane >> 4) * 8;
    v16bf afr[2], bfr[4];
#pragma unroll
    for (int mi = 0; mi < 2; ++mi) {
      const unsigned short* p = &lA[buf][(wm + mi * 16 + rc) * BK];
      v8bf lo = *(const v8bf*)(p + h8);
      v8bf hh = *(const v8bf*)(p + 16 + h8);
      afr[mi] = __builtin_shufflevector(lo, hh, 0,1,2,3,4,5,6,7,8,9,10,11,12,13,14,15);
    }
#pragma unroll
    for (int ni = 0; ni < 4; ++ni) {
      const unsigned short* p = &lB[buf][(wn + ni * 16 + rc) * BK];
      v8bf lo = *(const v8bf*)(p + h8);
      v8bf hh = *(const v8bf*)(p + 16 + h8);
      bfr[ni] = __builtin_shufflevector(lo, hh, 0,1,2,3,4,5,6,7,8,9,10,11,12,13,14,15);
    }
#pragma unroll
    for (int ni = 0; ni < 4; ++ni)
#pragma unroll
      for (int mi = 0; mi < 2; ++mi)
        acc[mi][ni] = __builtin_amdgcn_wmma_f32_16x16x32_bf16(
            false, afr[mi], false, bfr[ni], (short)0, acc[mi][ni], false, false);

    if (more) { storeA(buf ^ 1); storeB(buf ^ 1); }
  }

  // ---- epilogue: C/D layout: VGPR r -> M = r + (lane>>4)*8, N = lane&15 ----
#pragma unroll
  for (int mi = 0; mi < 2; ++mi)
#pragma unroll
    for (int ni = 0; ni < 4; ++ni)
#pragma unroll
      for (int r = 0; r < 8; ++r) {
        int gm = bm + wm + mi * 16 + (lane >> 4) * 8 + r;   // M % BM == 0: no guard
        int gn = bn + wn + ni * 16 + (lane & 15);
        if (gn < N) {
          float v = acc[mi][ni][r] + bias[gn];
          if (beta) v += C[(size_t)gm * ldc + gn];
          if (act == 1)      v = fmaxf(v, 0.f);
          else if (act == 2) v = rintf(fminf(fmaxf(v, 0.f), 1.f) + 0.1f); // round(clip+0.1)
          C[(size_t)gm * ldc + gn] = v;
        }
      }
}

// ---------------- conversion / utility kernels ------------------------------
__global__ void k_fill(float* __restrict__ p, size_t n, float v) {
  size_t i = (size_t)blockIdx.x * blockDim.x + threadIdx.x;
  if (i < n) p[i] = v;
}

// f32 [K][Nn] -> bf16 [Nn][K] (32x32 LDS tile transpose + convert), dims % 32 == 0
__global__ __launch_bounds__(256) void k_transconv(const float* __restrict__ in,
    unsigned short* __restrict__ out, int K, int Nn) {
  __shared__ unsigned short t[32][33];
  int n0 = blockIdx.x * 32, k0 = blockIdx.y * 32;
  int tx = threadIdx.x, ty = threadIdx.y;     // 32 x 8
#pragma unroll
  for (int j = 0; j < 4; ++j)
    t[ty + j * 8][tx] = f2bf(in[(size_t)(k0 + ty + j * 8) * Nn + (n0 + tx)]);
  __syncthreads();
#pragma unroll
  for (int j = 0; j < 4; ++j)
    out[(size_t)(n0 + ty + j * 8) * K + (k0 + tx)] = t[tx][ty + j * 8];
}

// u8 view-bitmask -> bf16 popcount values, 4 bytes/thread
__global__ void k_mask2bf(const unsigned* __restrict__ m, unsigned* __restrict__ o, size_t nw) {
  size_t i = (size_t)blockIdx.x * blockDim.x + threadIdx.x;
  if (i >= nw) return;
  unsigned b = m[i];
  o[i * 2 + 0] = pack2(f2bf((float)__popc((int)(b & 7u))),
                       f2bf((float)__popc((int)((b >> 8) & 7u))));
  o[i * 2 + 1] = pack2(f2bf((float)__popc((int)((b >> 16) & 7u))),
                       f2bf((float)__popc((int)((b >> 24) & 7u))));
}

__global__ void k_deg(const int* __restrict__ s, const int* __restrict__ d,
                      float* __restrict__ dout, float* __restrict__ din, int E) {
  int i = blockIdx.x * blockDim.x + threadIdx.x;
  if (i < E) { atomicAdd(&dout[s[i]], 1.f); atomicAdd(&din[d[i]], 1.f); }  // exact ints
}

__global__ void k_isqrt(float* __restrict__ p, int n) {
  int i = blockIdx.x * blockDim.x + threadIdx.x;
  if (i < n) p[i] = rsqrtf(fmaxf(p[i], 1.f));
}

__global__ void k_scatter(const float* __restrict__ h, const int* __restrict__ s,
                          const int* __restrict__ d, float* __restrict__ agg,
                          int E, int logF) {
  int i = blockIdx.x * blockDim.x + threadIdx.x;
  if (i >= (E << logF)) return;
  int e = i >> logF, c = i & ((1 << logF) - 1);
  atomicAdd(&agg[((size_t)d[e] << logF) + c], h[((size_t)s[e] << logF) + c]);
}

__global__ void k_fin(const float* __restrict__ agg, const float* __restrict__ di,
                      const float* __restrict__ b, float* __restrict__ out,
                      int n, int logF, int relu) {
  int i = blockIdx.x * blockDim.x + threadIdx.x;
  if (i >= n) return;
  int row = i >> logF, c = i & ((1 << logF) - 1);
  float v = agg[i] * di[row] + b[c];
  out[i] = relu ? fmaxf(v, 0.f) : v;
}

__global__ __launch_bounds__(256) void k_softmax_relu(float* __restrict__ z, int rows) {
  int lane = threadIdx.x & 31;
  int row = blockIdx.x * 8 + (threadIdx.x >> 5);
  if (row >= rows) return;
  float* p = z + (size_t)row * 64;
  float a = p[lane], b = p[lane + 32];
  float m = fmaxf(a, b);
#pragma unroll
  for (int s = 16; s; s >>= 1) m = fmaxf(m, __shfl_xor(m, s, 32));
  float ea = expf(a - m), eb = expf(b - m);
  float sum = ea + eb;
#pragma unroll
  for (int s = 16; s; s >>= 1) sum += __shfl_xor(sum, s, 32);
  float inv = 1.f / sum;
  p[lane]      = fmaxf(ea * inv, 0.f);
  p[lane + 32] = fmaxf(eb * inv, 0.f);
}

__global__ void k_adj_or(const int* __restrict__ s, const int* __restrict__ d,
                         unsigned* __restrict__ m, int E, int view) {
  int i = blockIdx.x * blockDim.x + threadIdx.x;
  if (i >= E) return;
  size_t idx = (size_t)s[i] * (size_t)GNODES + (size_t)(unsigned)d[i];
  atomicOr(&m[idx >> 2], (1u << view) << ((unsigned)(idx & 3u) * 8u));
}

// A[i][j] = (adj_r[i][j]>0 || adj_r[j][i]>0 || i==j) as bf16 {0,1}; dn[i]=deg^-1/2
__global__ __launch_bounds__(128) void k_buildA(const float* __restrict__ adjr,
                                                unsigned short* __restrict__ A,
                                                float* __restrict__ dn) {
  __shared__ int red[128];
  int i = blockIdx.x, t = threadIdx.x;
  int cnt = 0;
  for (int j = t; j < GNODES; j += 128) {
    bool a = (adjr[(size_t)i * GNODES + j] != 0.f) ||
             (adjr[(size_t)j * GNODES + i] != 0.f) || (i == j);
    A[(size_t)i * GNODES + j] = a ? (unsigned short)0x3F80 : (unsigned short)0;
    cnt += a ? 1 : 0;
  }
  red[t] = cnt; __syncthreads();
  for (int s = 64; s; s >>= 1) { if (t < s) red[t] += red[t + s]; __syncthreads(); }
  if (t == 0) dn[i] = rsqrtf((float)red[0]);   // deg >= 1 (self-loop)
}

// ---------------- host-side launchers ---------------------------------------
static void gemm(hipStream_t st, const void* A, const void* B, float* C,
                 int N, int K, int lda, int ldb, int ldc,
                 const float* ar, const float* br, const float* bias,
                 int amode, int bmode, int beta, int act) {
  dim3 g((unsigned)((N + BN - 1) / BN), (unsigned)(GNODES / BM)), blk(256);
  if      (amode == 0 && bmode == 0) gemm_wmma<0,0><<<g,blk,0,st>>>(A,B,C,N,K,lda,ldb,ldc,ar,br,bias,beta,act);
  else if (amode == 1 && bmode == 0) gemm_wmma<1,0><<<g,blk,0,st>>>(A,B,C,N,K,lda,ldb,ldc,ar,br,bias,beta,act);
  else if (amode == 2 && bmode == 0) gemm_wmma<2,0><<<g,blk,0,st>>>(A,B,C,N,K,lda,ldb,ldc,ar,br,bias,beta,act);
  else if (amode == 0 && bmode == 1) gemm_wmma<0,1><<<g,blk,0,st>>>(A,B,C,N,K,lda,ldb,ldc,ar,br,bias,beta,act);
  else if (amode == 2 && bmode == 2) gemm_wmma<2,2><<<g,blk,0,st>>>(A,B,C,N,K,lda,ldb,ldc,ar,br,bias,beta,act);
  else                               gemm_wmma<0,2><<<g,blk,0,st>>>(A,B,C,N,K,lda,ldb,ldc,ar,br,bias,beta,act);
}
static void fill(hipStream_t st, float* p, size_t n, float v) {
  k_fill<<<(unsigned)((n + 255) / 256), 256, 0, st>>>(p, n, v);
}

extern "C" void kernel_launch(void* const* d_in, const int* in_sizes, int n_in,
                              void* d_out, int out_size, void* d_ws, size_t ws_size,
                              hipStream_t stream) {
  (void)in_sizes; (void)n_in; (void)out_size;

  const float *x[3], *Wv0[3], *bv0[3], *Wv1[3], *bv1[3], *Wf[3];
  const int *srcv[3], *dstv[3];
  for (int v = 0; v < 3; ++v) {
    int b = v * 8;
    x[v]   = (const float*)d_in[b + 0];
    srcv[v]= (const int*)  d_in[b + 1];
    dstv[v]= (const int*)  d_in[b + 2];
    Wv0[v] = (const float*)d_in[b + 3];
    bv0[v] = (const float*)d_in[b + 4];
    Wv1[v] = (const float*)d_in[b + 5];
    bv1[v] = (const float*)d_in[b + 6];
    Wf[v]  = (const float*)d_in[b + 7];
  }
  const float* Wg1 = (const float*)d_in[24]; const float* bg1 = (const float*)d_in[25];
  const float* Wg2 = (const float*)d_in[26]; const float* bg2 = (const float*)d_in[27];
  const float* Wm0 = (const float*)d_in[28]; const float* bm0 = (const float*)d_in[29];
  const float* Wm1 = (const float*)d_in[30]; const float* bm1 = (const float*)d_in[31];

  // ---- workspace layout (floats) ----
  float* ws = (float*)d_ws;
  const size_t OF_DNORM = 0;                         // 6*4096
  const size_t OF_H1    = OF_DNORM + 6 * GNODES;     // 4096*128
  const size_t OF_AGG   = OF_H1 + (size_t)GNODES * GH0;
  const size_t OF_F     = OF_AGG + (size_t)GNODES * GH0;   // 3 * 4096*64
  const size_t OF_Z     = OF_F + 3 * (size_t)GNODES * GH1;
  const size_t OF_HMID  = OF_Z + (size_t)GNODES * GH1;
  const size_t OF_DN    = OF_HMID + (size_t)GNODES * GH1;
  const size_t OF_ONES  = OF_DN + GNODES;
  const size_t OF_ZEROS = OF_ONES + GNODES;
  const size_t OF_BIG   = OF_ZEROS + GNODES;
  const size_t QNN      = (size_t)GNODES * GNODES / 4;      // 4Mi floats (u8 N*N)
  const size_t HNN      = (size_t)GNODES * GNODES / 2;      // 8.4Mi floats (bf16 N*N / f32 N*NH)
  // small-ws path: big = [adjmask QNN][C1 HNN]; fused-A bf16 overlays big
  // big-ws  path: [R0 adjmask QNN][R1 HNN: Adjbf->Wg2bf->fusedA][R2 HNN: C1][R3 QNN: Wg1bf]
  const size_t NEED_BIG = (OF_BIG + QNN + HNN + HNN + QNN) * 4;
  const bool   big      = ws_size >= NEED_BIG;

  float* dnorm = ws + OF_DNORM;
  float* h1b   = ws + OF_H1;
  float* agg   = ws + OF_AGG;
  float* fb    = ws + OF_F;
  float* z     = ws + OF_Z;
  float* hmid  = ws + OF_HMID;
  float* dn    = ws + OF_DN;
  float* ones  = ws + OF_ONES;
  float* zeros = ws + OF_ZEROS;
  float* R0 = ws + OF_BIG;            // adjmask (u8 bitmask)
  float* R1 = R0 + QNN;
  float* R2 = R1 + HNN;
  float* R3 = R2 + HNN;
  unsigned* adjmask = (unsigned*)R0;
  float* C1                  = big ? R2 : R1;
  unsigned short* Adjbf      = (unsigned short*)R1;   // big path only
  unsigned short* Wg1bf      = (unsigned short*)R3;   // big path only
  unsigned short* Wg2bf      = (unsigned short*)R1;   // big path, after GEMM1
  unsigned short* Abf        = big ? (unsigned short*)R1 : (unsigned short*)R0;

  float* out_adjr = (float*)d_out;                       // [N,N]
  float* out_rec  = out_adjr + (size_t)GNODES * GNODES;  // [N,N]
  float* out_h    = out_rec  + (size_t)GNODES * GNODES;  // [N,64]

  // ---- constants + degree norms ----
  fill(stream, ones,  GNODES, 1.f);
  fill(stream, zeros, GNODES, 0.f);
  fill(stream, dnorm, 6 * GNODES, 0.f);
  for (int v = 0; v < 3; ++v)
    k_deg<<<(GEDGES + 255) / 256, 256, 0, stream>>>(srcv[v], dstv[v],
        dnorm + v * 2 * GNODES, dnorm + v * 2 * GNODES + GNODES, GEDGES);
  k_isqrt<<<(6 * GNODES + 255) / 256, 256, 0, stream>>>(dnorm, 6 * GNODES);

  // ---- per-view 2-layer GCN ----
  for (int v = 0; v < 3; ++v) {
    const float* doV = dnorm + v * 2 * GNODES;
    const float* diV = doV + GNODES;
    float* fv = fb + (size_t)v * GNODES * GH1;
    gemm(stream, x[v], Wv0[v], h1b, GH0, GIN, GIN, GH0, GH0, doV, ones, zeros, 0, 0, 0, 0);
    fill(stream, agg, (size_t)GNODES * GH0, 0.f);
    k_scatter<<<(GEDGES << 7) / 256, 256, 0, stream>>>(h1b, srcv[v], dstv[v], agg, GEDGES, 7);
    k_fin<<<((GNODES * GH0) + 255) / 256, 256, 0, stream>>>(agg, diV, bv0[v], h1b,
        GNODES * GH0, 7, 1);
    gemm(stream, h1b, Wv1[v], fv, GH1, GH0, GH0, GH1, GH1, doV, ones, zeros, 0, 0, 0, 0);
    fill(stream, agg, (size_t)GNODES * GH1, 0.f);
    k_scatter<<<(GEDGES << 6) / 256, 256, 0, stream>>>(fv, srcv[v], dstv[v], agg, GEDGES, 6);
    k_fin<<<((GNODES * GH1) + 255) / 256, 256, 0, stream>>>(agg, diV, bv1[v], fv,
        GNODES * GH1, 6, 0);
  }

  // ---- feature fusion: z = sum_v f_v @ Wf_v, then relu(softmax(z)) ----
  for (int v = 0; v < 3; ++v)
    gemm(stream, fb + (size_t)v * GNODES * GH1, Wf[v], z, GH1, GH1,
         GH1, GH1, GH1, ones, ones, zeros, 0, 0, v ? 1 : 0, 0);
  k_softmax_relu<<<GNODES / 8, 256, 0, stream>>>(z, GNODES);

  // ---- consensus adjacency; GFN (dominant 137 GFLOP) ----
  fill(stream, R0, QNN, 0.f);
  for (int v = 0; v < 3; ++v)
    k_adj_or<<<(GEDGES + 255) / 256, 256, 0, stream>>>(srcv[v], dstv[v], adjmask, GEDGES, v);
  if (big) {   // pre-converted bf16 operands: zero-conversion staging in the hot GEMMs
    k_mask2bf<<<(unsigned)(QNN / 256), 256, 0, stream>>>(adjmask, (unsigned*)Adjbf, QNN);
    k_transconv<<<dim3(GNH / 32, GNODES / 32), dim3(32, 8), 0, stream>>>(Wg1, Wg1bf, GNODES, GNH);
    gemm(stream, Adjbf, Wg1bf, C1, GNH, GNODES, GNODES, GNODES, GNH,
         ones, ones, bg1, 2, 2, 0, 1);
    k_transconv<<<dim3(GNODES / 32, GNH / 32), dim3(32, 8), 0, stream>>>(Wg2, Wg2bf, GNH, GNODES);
    gemm(stream, C1, Wg2bf, out_adjr, GNODES, GNH, GNH, GNH, GNODES,
         ones, ones, bg2, 0, 2, 0, 2);
  } else {     // compact path: popcount A staging + f32->bf16 B staging
    gemm(stream, adjmask, Wg1, C1, GNH, GNODES, GNODES, GNH, GNH,
         ones, ones, bg1, 1, 0, 0, 1);
    gemm(stream, C1, Wg2, out_adjr, GNODES, GNH, GNH, GNODES, GNODES,
         ones, ones, bg2, 0, 0, 0, 2);
  }

  // ---- rebuild symmetric binary graph + self-loops, deg norms ----
  k_buildA<<<GNODES, 128, 0, stream>>>(out_adjr, Abf, dn);

  // ---- fused-graph GCN ----
  gemm(stream, Abf, z, agg, GH1, GNODES, GNODES, GH1, GH1, ones, dn, zeros, 2, 0, 0, 0);
  gemm(stream, agg, Wm0, hmid, GH1, GH1, GH1, GH1, GH1, dn, ones, bm0, 0, 0, 0, 1);
  gemm(stream, Abf, hmid, agg, GH1, GNODES, GNODES, GH1, GH1, ones, dn, zeros, 2, 0, 0, 0);
  gemm(stream, agg, Wm1, out_h, GH1, GH1, GH1, GH1, GH1, dn, ones, bm1, 0, 0, 0, 0);

  // ---- decoder: adj_rec = h @ h.T (B-transposed GEMM) ----
  gemm(stream, out_h, out_h, out_rec, GNODES, GH1, GH1, GH1, GNODES,
       ones, ones, zeros, 0, 1, 0, 0);
}
